// Engram_6536940225178
// MI455X (gfx1250) — matrix operand order repaired and checked by reading, not compiled, not run on hardware
//
#include <hip/hip_runtime.h>
#include <hip/hip_bf16.h>

typedef __attribute__((ext_vector_type(16))) __bf16 v16bf;
typedef __attribute__((ext_vector_type(8)))  float  v8f;

#define TN   2048
#define GN   4
#define HN   1024
#define NHD  16
#define DHD  64
#define BTN  8192            // B*T
#define KDIM 1024            // contraction length (e)
#define NTOT 5120            // value(1024) + 4*keys(1024)

struct OffsArg { long long o[16]; };

// ---------------------------------------------------------------------------
// block-wide sum reduction (256 threads = 8 wave32)
// ---------------------------------------------------------------------------
__device__ __forceinline__ float block_reduce_sum(float v, float* sm) {
#pragma unroll
    for (int o = 16; o > 0; o >>= 1) v += __shfl_down(v, o, 32);
    const int lane = threadIdx.x & 31;
    const int w    = threadIdx.x >> 5;
    if (lane == 0) sm[w] = v;
    __syncthreads();
    if (threadIdx.x < 32) {
        float t = (lane < 8) ? sm[lane] : 0.f;
#pragma unroll
        for (int o = 4; o > 0; o >>= 1) t += __shfl_down(t, o, 32);
        if (lane == 0) sm[0] = t;
    }
    __syncthreads();
    float r = sm[0];
    __syncthreads();
    return r;
}

// ---------------------------------------------------------------------------
// K1: concat value_w (1M) + key_w (4M) and cast f32 -> bf16   ([n][e] layout)
// ---------------------------------------------------------------------------
__global__ __launch_bounds__(256) void prep_weights_kernel(
        const float* __restrict__ value_w, const float* __restrict__ key_w,
        __bf16* __restrict__ wcat) {
    const size_t i  = (size_t)blockIdx.x * 256 + threadIdx.x;
    const size_t nv = (size_t)HN * KDIM;                 // 1M
    const size_t nt = (size_t)NTOT * KDIM;               // 5M
    if (i >= nt) return;
    float v = (i < nv) ? value_w[i] : key_w[i - nv];
    wcat[i] = (__bf16)v;
}

// K1b: concat value_b (1024) + key_b (4096) -> biascat[5120]
__global__ __launch_bounds__(256) void prep_bias_kernel(
        const float* __restrict__ value_b, const float* __restrict__ key_b,
        float* __restrict__ biascat) {
    const int i = blockIdx.x * 256 + threadIdx.x;
    if (i >= NTOT) return;
    biascat[i] = (i < HN) ? value_b[i] : key_b[i - HN];
}

// ---------------------------------------------------------------------------
// K2: gather embeddings -> bf16 A matrix [BTN][1024]
// ---------------------------------------------------------------------------
__global__ __launch_bounds__(256) void gather_kernel(
        const float* __restrict__ emb, const int* __restrict__ hash,
        OffsArg offs, __bf16* __restrict__ out) {
    const size_t idx = (size_t)blockIdx.x * 256 + threadIdx.x;   // < BTN*1024
    const int    d   = (int)(idx & 63);
    const int    h   = (int)((idx >> 6) & 15);
    const size_t bt  = idx >> 10;
    const long long row = (long long)hash[bt * NHD + h] + offs.o[h];
    out[idx] = (__bf16)emb[(size_t)row * DHD + d];
}

// ---------------------------------------------------------------------------
// K3: xn = rmsnorm(hidden_states, sc_norm_w, eps=1e-5)   one block per row
// ---------------------------------------------------------------------------
__global__ __launch_bounds__(256) void rmsnorm_sc_kernel(
        const float* __restrict__ hs, const float* __restrict__ scw,
        float* __restrict__ xn) {
    __shared__ float sm[8];
    const int row = blockIdx.x;                    // (b*T + t)*G + g
    const int g   = row & 3;
    const int h0  = threadIdx.x * 4;
    const float4 xv = *(const float4*)(hs + (size_t)row * HN + h0);
    float ss = xv.x*xv.x + xv.y*xv.y + xv.z*xv.z + xv.w*xv.w;
    const float tot = block_reduce_sum(ss, sm);
    const float rs  = rsqrtf(tot * (1.f / HN) + 1e-5f);
    const float4 wv = *(const float4*)(scw + g * HN + h0);
    float4 o;
    o.x = xv.x * rs * wv.x;  o.y = xv.y * rs * wv.y;
    o.z = xv.z * rs * wv.z;  o.w = xv.w * rs * wv.w;
    *(float4*)(xn + (size_t)row * HN + h0) = o;
}

// ---------------------------------------------------------------------------
// K4: causal dilated depthwise conv (K=4, dil=3) + SiLU + rmsnorm -> q
// ---------------------------------------------------------------------------
__global__ __launch_bounds__(256) void conv_q_kernel(
        const float* __restrict__ xn, const float* __restrict__ convw,
        const float* __restrict__ n1w, float* __restrict__ q) {
    __shared__ float sm[8];
    const int row = blockIdx.x;                    // bt*G + g
    const int g   = row & 3;
    const int t   = (row >> 2) & (TN - 1);
    const int h0  = threadIdx.x * 4;
    float acc[4] = {0.f, 0.f, 0.f, 0.f};
    // taps are contiguous in conv_w: [(g*HN+h)*4 + k]
    float4 w0 = *(const float4*)(convw + ((size_t)(g * HN + h0 + 0)) * 4);
    float4 w1 = *(const float4*)(convw + ((size_t)(g * HN + h0 + 1)) * 4);
    float4 w2 = *(const float4*)(convw + ((size_t)(g * HN + h0 + 2)) * 4);
    float4 w3 = *(const float4*)(convw + ((size_t)(g * HN + h0 + 3)) * 4);
    const float* wk[4] = {(const float*)&w0, (const float*)&w1,
                          (const float*)&w2, (const float*)&w3};
#pragma unroll
    for (int k = 0; k < 4; ++k) {
        const int tt = t - 9 + 3 * k;
        if (tt >= 0) {
            const float4 xv = *(const float4*)(xn +
                ((size_t)row - (size_t)(9 - 3 * k) * GN) * HN + h0);
            acc[0] += xv.x * wk[0][k];
            acc[1] += xv.y * wk[1][k];
            acc[2] += xv.z * wk[2][k];
            acc[3] += xv.w * wk[3][k];
        }
    }
    float y[4], ss = 0.f;
#pragma unroll
    for (int j = 0; j < 4; ++j) {
        const float v = acc[j];
        y[j] = v / (1.f + expf(-v));               // silu
        ss += y[j] * y[j];
    }
    const float tot = block_reduce_sum(ss, sm);
    const float rs  = rsqrtf(tot * (1.f / HN) + 1e-6f);
    const float* nw = n1w + g * HN + h0;
    float4 o;
    o.x = y[0] * rs * nw[0];  o.y = y[1] * rs * nw[1];
    o.z = y[2] * rs * nw[2];  o.w = y[3] * rs * nw[3];
    *(float4*)(q + (size_t)row * HN + h0) = o;
}

// ---------------------------------------------------------------------------
// K5: WMMA GEMM  C[8192][5120] = embs_bf16 @ wcatT + biascat
//     block = 8 waves -> 128(M) x 256(N); wave -> 64(M) x 64(N) = 16 tiles
//     (16 wmma per 16 b128 loads per K-step: ~2x reuse vs 32x64 tiles)
// ---------------------------------------------------------------------------
__global__ __launch_bounds__(256) void gemm_wmma_kernel(
        const __bf16* __restrict__ A,      // [BTN][KDIM]
        const __bf16* __restrict__ Wc,     // [NTOT][KDIM]
        const float*  __restrict__ biasc,  // [NTOT]
        float* __restrict__ C) {           // [BTN][NTOT]
    const int w    = threadIdx.x >> 5;
    const int lane = threadIdx.x & 31;
    const int hi   = lane >> 4;            // half-wave index
    const int lo   = lane & 15;
    const int m_wave = blockIdx.y * 128 + (w & 1) * 64;
    const int n_wave = blockIdx.x * 256 + (w >> 1) * 64;

    // fragment base pointers (ISA 16-bit A 16x32 layout: lanes 0-15 K={0..7,16..23},
    // lanes 16-31 K={8..15,24..31}; B: same K split per lane-column)
    const __bf16* pa[4];
#pragma unroll
    for (int mt = 0; mt < 4; ++mt)
        pa[mt] = A + (size_t)(m_wave + mt * 16 + lo) * KDIM + hi * 8;
    const __bf16* pb[4];
#pragma unroll
    for (int nt = 0; nt < 4; ++nt)
        pb[nt] = Wc + (size_t)(n_wave + nt * 16 + lo) * KDIM + hi * 8;

    v8f acc[4][4] = {};
    for (int kb = 0; kb < KDIM; kb += 32) {
        v16bf a[4], b[4];
#pragma unroll
        for (int mt = 0; mt < 4; ++mt) {
            union { v16bf v; float4 f[2]; } u;
            u.f[0] = *(const float4*)(pa[mt] + kb);        // K 0..7   (lo half)
            u.f[1] = *(const float4*)(pa[mt] + kb + 16);   // K 16..23 (lo half)
            a[mt] = u.v;
        }
#pragma unroll
        for (int nt = 0; nt < 4; ++nt) {
            union { v16bf v; float4 f[2]; } u;
            u.f[0] = *(const float4*)(pb[nt] + kb);
            u.f[1] = *(const float4*)(pb[nt] + kb + 16);
            b[nt] = u.v;
        }
        // speculative prefetch of next K-slice (locality 3 -> near scope);
        // overshoot past the row is silently dropped
        __builtin_prefetch(pa[0] + kb + 32, 0, 3);
        __builtin_prefetch(pb[0] + kb + 32, 0, 3);
#pragma unroll
        for (int mt = 0; mt < 4; ++mt)
#pragma unroll
            for (int nt = 0; nt < 4; ++nt)
                acc[mt][nt] = __builtin_amdgcn_wmma_f32_16x16x32_bf16(
                    false, a[mt], false, b[nt], (short)0, acc[mt][nt],
                    false, false);
    }

    // C/D layout: VGPR r, lanes 0-15 -> M=r, N=lane; lanes 16-31 -> M=r+8
#pragma unroll
    for (int mt = 0; mt < 4; ++mt)
#pragma unroll
        for (int nt = 0; nt < 4; ++nt) {
            const int col  = n_wave + nt * 16 + lo;
            const float bs = biasc[col];
#pragma unroll
            for (int r = 0; r < 8; ++r) {
                const int row = m_wave + mt * 16 + hi * 8 + r;
                C[(size_t)row * NTOT + col] = acc[mt][nt][r] + bs;
            }
        }
}

// ---------------------------------------------------------------------------
// K6: k = rmsnorm(keys, norm2_w); gate = sigmoid(q.k / 32); out = gate*value
// ---------------------------------------------------------------------------
__global__ __launch_bounds__(256) void final_kernel(
        const float* __restrict__ Cmat, const float* __restrict__ q,
        const float* __restrict__ n2w, float* __restrict__ out) {
    __shared__ float sm[8];
    const int row = blockIdx.x;                    // bt*G + g
    const int g   = row & 3;
    const size_t bt = (size_t)(row >> 2);
    const int h0  = threadIdx.x * 4;
    const float* keys = Cmat + bt * NTOT + HN + (size_t)g * HN;
    const float* val  = Cmat + bt * NTOT;

    const float4 kv = *(const float4*)(keys + h0);
    float ss = kv.x*kv.x + kv.y*kv.y + kv.z*kv.z + kv.w*kv.w;
    const float kt = block_reduce_sum(ss, sm);
    const float rk = rsqrtf(kt * (1.f / HN) + 1e-6f);

    const float4 qv = *(const float4*)(q + (size_t)row * HN + h0);
    const float4 wv = *(const float4*)(n2w + g * HN + h0);
    float dpart = qv.x*kv.x*wv.x + qv.y*kv.y*wv.y +
                  qv.z*kv.z*wv.z + qv.w*kv.w*wv.w;
    const float dtot = block_reduce_sum(dpart, sm);
    const float dot  = dtot * rk;
    const float gate = 1.f / (1.f + expf(-dot * (1.f / 32.f)));  // sqrt(1024)=32

    const float4 vv = *(const float4*)(val + h0);
    float4 o;
    o.x = gate * vv.x;  o.y = gate * vv.y;
    o.z = gate * vv.z;  o.w = gate * vv.w;
    *(float4*)(out + (size_t)row * HN + h0) = o;
}

// ---------------------------------------------------------------------------
// host side
// ---------------------------------------------------------------------------
static bool isprime_h(long long n) {
    if (n < 2) return false;
    if (n % 2 == 0) return n == 2;
    for (long long i = 3; i * i <= n; i += 2)
        if (n % i == 0) return false;
    return true;
}

extern "C" void kernel_launch(void* const* d_in, const int* in_sizes, int n_in,
                              void* d_out, int out_size, void* d_ws, size_t ws_size,
                              hipStream_t stream) {
    (void)in_sizes; (void)n_in; (void)out_size; (void)ws_size;
    const float* hidden   = (const float*)d_in[0];
    const int*   hash_ids = (const int*)  d_in[1];
    const float* emb_tab  = (const float*)d_in[2];
    const float* conv_w   = (const float*)d_in[3];
    const float* sc_norm  = (const float*)d_in[4];
    const float* value_w  = (const float*)d_in[5];
    const float* value_b  = (const float*)d_in[6];
    const float* key_w    = (const float*)d_in[7];
    const float* key_b    = (const float*)d_in[8];
    const float* norm1_w  = (const float*)d_in[9];
    const float* norm2_w  = (const float*)d_in[10];
    float* outp = (float*)d_out;

    // deterministic host-side offset computation (primes >= 129280)
    OffsArg offs;
    {
        long long list[16], seen[16];
        int ns = 0, cnt = 0;
        for (int vi = 0; vi < 2; ++vi) {
            long long start = 129280 - 1;
            for (int hh = 0; hh < 8; ++hh) {
                long long c = start + 1;
                for (;;) {
                    bool ok = isprime_h(c);
                    if (ok)
                        for (int s = 0; s < ns; ++s)
                            if (seen[s] == c) { ok = false; break; }
                    if (ok) break;
                    ++c;
                }
                seen[ns++] = c; list[cnt++] = c; start = c;
            }
        }
        long long cum = 0;
        for (int i = 0; i < 16; ++i) { offs.o[i] = cum; cum += list[i]; }
    }

    // workspace layout (bytes)
    char* ws = (char*)d_ws;
    __bf16* wcat    = (__bf16*)(ws + 0);                    // 10,485,760
    float*  biascat = (float*) (ws + 10485760);             //     20,480
    __bf16* embs    = (__bf16*)(ws + 10506240);             // 16,777,216
    float*  xn      = (float*) (ws + 27283456);             // 134,217,728
    float*  qbuf    = (float*) (ws + 161501184);            // 134,217,728
    float*  Cmat    = (float*) (ws + 295718912);            // 167,772,160

    prep_weights_kernel<<<dim3((NTOT * KDIM + 255) / 256), dim3(256), 0, stream>>>(
        value_w, key_w, wcat);
    prep_bias_kernel<<<dim3((NTOT + 255) / 256), dim3(256), 0, stream>>>(
        value_b, key_b, biascat);
    gather_kernel<<<dim3(BTN * KDIM / 256), dim3(256), 0, stream>>>(
        emb_tab, hash_ids, offs, embs);
    rmsnorm_sc_kernel<<<dim3(BTN * GN), dim3(256), 0, stream>>>(
        hidden, sc_norm, xn);
    conv_q_kernel<<<dim3(BTN * GN), dim3(256), 0, stream>>>(
        xn, conv_w, norm1_w, qbuf);
    gemm_wmma_kernel<<<dim3(NTOT / 256, BTN / 128), dim3(256), 0, stream>>>(
        embs, wcat, biascat, Cmat);
    final_kernel<<<dim3(BTN * GN), dim3(256), 0, stream>>>(
        Cmat, qbuf, norm2_w, outp);
}
